// MultiHeadSelfAttention_10015863734457
// MI455X (gfx1250) — matrix operand order repaired and checked by reading, not compiled, not run on hardware
//
#include <hip/hip_runtime.h>

// ---------------------------------------------------------------------------
// MHSA forward for MI455X (gfx1250): wave32, v_wmma_f32_16x16x32_f16 core,
// flash-attention with double-buffered K/V staging into LDS via
// global_load_async_to_lds_b128 (ASYNCcnt path), shared across 4 waves/WG.
// ---------------------------------------------------------------------------

typedef _Float16 v8h  __attribute__((ext_vector_type(8)));
typedef _Float16 v16h __attribute__((ext_vector_type(16)));
typedef float    v8f  __attribute__((ext_vector_type(8)));

#define D_MODEL  1024
#define N_HEADS  16
#define HEAD_DIM 64
#define SEQ      2048
#define BATCH    4
#define NTOK     (BATCH * SEQ)          // 8192 tokens
#define ATT_SCALE 0.125f                // 1/sqrt(64)

// D = A(16x32 f16) * B(32x16 f16) + C(16x16 f32)
static __device__ __forceinline__ v8f wmma_f16(v16h a, v16h b, v8f c) {
  return __builtin_amdgcn_wmma_f32_16x16x32_f16(false, a, false, b,
                                                (short)0, c, false, false);
}

// A-matrix per-lane fragment (contiguous k order used by this file, same as load_b16):
// lane L holds row M=L&15, K = kb..kb+15 with kb=(L>>4)*16.
// Caller passes p = row_ptr + ktile + (L>>4)*16.
static __device__ __forceinline__ v16h load_a16(const _Float16* p) {
  v8h lo = *(const v8h*)p;
  v8h hi = *(const v8h*)(p + 8);
  v16h r;
#pragma unroll
  for (int i = 0; i < 8; ++i) { r[i] = lo[i]; r[i + 8] = hi[i]; }
  return r;
}

// B-matrix per-lane fragment (32x16): lane L holds col N=L&15,
// K = (L>>4)*16 + j, j=0..15 -> 16 contiguous halfs (32B aligned).
static __device__ __forceinline__ v16h load_b16(const _Float16* p) {
  return *(const v16h*)p;
}

// --- CDNA5 async global->LDS copy (ASYNCcnt path, ISA ch.10 / 08_async) ----
// LDS destination is the low 32 bits of the generic pointer (LDS aperture
// addresses truncate to the wave-relative LDS byte offset, ISA 10.2).
static __device__ __forceinline__ unsigned lds_off(const void* p) {
  return (unsigned)(size_t)p;
}
static __device__ __forceinline__ void async_copy_b128(unsigned lds_dst,
                                                       const void* gsrc) {
  asm volatile("global_load_async_to_lds_b128 %0, %1, off"
               :: "v"(lds_dst), "v"(gsrc) : "memory");
}
// Wait until ASYNCcnt <= N.  Async loads complete in order, so N=4 releases
// the older of two 4-op staging batches while the newer stays in flight.
template <int N>
static __device__ __forceinline__ void wait_async() {
  asm volatile("s_wait_asynccnt %0" :: "i"(N) : "memory");
}

// ---------------------------------------------------------------------------
// Kernel 1: fp32 -> f16 conversion (x and weight matrices)
// ---------------------------------------------------------------------------
__global__ void cvt_f32_to_f16(const float* __restrict__ src,
                               _Float16* __restrict__ dst, int n) {
  int i = blockIdx.x * blockDim.x + threadIdx.x;
  int stride = gridDim.x * blockDim.x;
  for (; i < n; i += stride) dst[i] = (_Float16)src[i];
}

// ---------------------------------------------------------------------------
// Kernel 2: fused QKV projection.  One wave -> one 16x16 output tile for each
// of Q, K, V (shared A fragment).  Y = X @ W^T + b.
// Q,K stored [B,H,S,64]; V stored transposed [B,H,64,S] so the attention
// P@V B-matrix loads are contiguous.
// ---------------------------------------------------------------------------
__global__ void __launch_bounds__(256)
qkv_gemm(const _Float16* __restrict__ Xh,
         const _Float16* __restrict__ Wqh,
         const _Float16* __restrict__ Wkh,
         const _Float16* __restrict__ Wvh,
         const float* __restrict__ bq,
         const float* __restrict__ bk,
         const float* __restrict__ bv,
         _Float16* __restrict__ Qh,
         _Float16* __restrict__ Kh,
         _Float16* __restrict__ Vt) {
  const int lane = threadIdx.x & 31;
  const int wid  = threadIdx.x >> 5;
  const int tile = blockIdx.x * 8 + wid;       // 32768 tiles = 512 m x 64 n
  const int m0 = (tile >> 6) * 16;
  const int n0 = (tile & 63) * 16;
  const int g = lane >> 4, ln = lane & 15;

  const _Float16* Arow = Xh  + (size_t)(m0 + ln) * D_MODEL + g * 16;
  const _Float16* Bq   = Wqh + (size_t)(n0 + ln) * D_MODEL + g * 16;
  const _Float16* Bk   = Wkh + (size_t)(n0 + ln) * D_MODEL + g * 16;
  const _Float16* Bv   = Wvh + (size_t)(n0 + ln) * D_MODEL + g * 16;

  v8f cq = {}, ck = {}, cv = {};
  for (int k0 = 0; k0 < D_MODEL; k0 += 32) {
    __builtin_prefetch(Arow + k0 + 128, 0, 3);
    v16h a  = load_a16(Arow + k0);
    v16h b0 = load_b16(Bq + k0);
    v16h b1 = load_b16(Bk + k0);
    v16h b2 = load_b16(Bv + k0);
    cq = wmma_f16(a, b0, cq);
    ck = wmma_f16(a, b1, ck);
    cv = wmma_f16(a, b2, cv);
  }

  const int e = n0 + ln;
  const int h = e >> 6, d = e & 63;
  const float biq = bq[e], bik = bk[e], biv = bv[e];
#pragma unroll
  for (int v = 0; v < 8; ++v) {
    const int m = m0 + g * 8 + v;              // global token index
    const int bb = m >> 11, s = m & (SEQ - 1);
    const size_t qi = (((size_t)bb * N_HEADS + h) * SEQ + s) * HEAD_DIM + d;
    Qh[qi] = (_Float16)(cq[v] + biq);
    Kh[qi] = (_Float16)(ck[v] + bik);
    const size_t vi = (((size_t)bb * N_HEADS + h) * HEAD_DIM + d) * SEQ + s;
    Vt[vi] = (_Float16)(cv[v] + biv);
  }
}

// ---------------------------------------------------------------------------
// Kernel 3: flash attention.  4 waves per workgroup share one (b,h).
// K/V 32-key blocks are staged into double-buffered LDS tiles with
// global_load_async_to_lds_b128; s_wait_asynccnt 4 overlaps the next block's
// copies with the current block's 8 WMMAs + online softmax.
// ---------------------------------------------------------------------------
__global__ void __launch_bounds__(128)
attn_kernel(const _Float16* __restrict__ Qh,
            const _Float16* __restrict__ Kh,
            const _Float16* __restrict__ Vt,
            _Float16* __restrict__ Ctx) {
  __shared__ __align__(32) _Float16 Klds[2][32 * HEAD_DIM];  // [key][d]  2x4KB
  __shared__ __align__(32) _Float16 Vlds[2][HEAD_DIM * 32];  // [d][key]  2x4KB
  __shared__ __align__(32) _Float16 Plds[4][16 * 32];        // per-wave P 4KB

  const int t    = threadIdx.x;                // 0..127
  const int lane = t & 31;
  const int w    = t >> 5;                     // wave id 0..3
  const int g = lane >> 4, ln = lane & 15;
  const int bh = blockIdx.x;                   // b*16 + h
  const int bb = bh >> 4, h = bh & 15;
  const int q0 = blockIdx.y * 64 + w * 16;     // this wave's query tile

  const _Float16* Qp = Qh + ((size_t)bh * SEQ + q0) * HEAD_DIM;
  const _Float16* Kp = Kh + (size_t)bh * SEQ * HEAD_DIM;
  const _Float16* Vp = Vt + (size_t)bh * HEAD_DIM * SEQ;

  const unsigned klb[2] = { lds_off(Klds[0]), lds_off(Klds[1]) };
  const unsigned vlb[2] = { lds_off(Vlds[0]), lds_off(Vlds[1]) };
  _Float16* Pw = Plds[w];

  // Issue one 32-key staging batch (4 async b128 per thread = 8 KB total).
  auto stage_kv = [&](int buf, int j) {
    const char* kq = (const char*)(Kp + (size_t)j * HEAD_DIM);
#pragma unroll
    for (int r = 0; r < 2; ++r) {
      const int c = t + r * 128;               // 256 x 16B chunks each array
      async_copy_b128(klb[buf] + c * 16, kq + c * 16);
      const int d = c >> 2, sub = c & 3;
      async_copy_b128(vlb[buf] + c * 16,
                      (const char*)(Vp + (size_t)d * SEQ + j) + sub * 16);
    }
  };

  // Q tile as two A fragments (head_dim 64 = 2 k-steps of 32)
  const v16h aq0 = load_a16(Qp + (size_t)ln * HEAD_DIM + g * 16);
  const v16h aq1 = load_a16(Qp + (size_t)ln * HEAD_DIM + 32 + g * 16);

  v8f acc0 = {}, acc1 = {}, acc2 = {}, acc3 = {};   // 16x64 ctx accumulator
  float Mr[8], Lr[8];
#pragma unroll
  for (int v = 0; v < 8; ++v) { Mr[v] = -1e30f; Lr[v] = 0.f; }

  stage_kv(0, 0);                               // prologue fill of buffer 0

  for (int j = 0; j < SEQ; j += 32) {
    const int cur = (j >> 5) & 1;
    // Prefetch next block into the other buffer, then release current buffer.
    if (j + 32 < SEQ) { stage_kv(cur ^ 1, j + 32); wait_async<4>(); }
    else              { wait_async<0>(); }
    __syncthreads();                            // current K/V visible to all

    // ---- scores for keys [j, j+32): two 16x16 f32 tiles from LDS K -------
    v8f s0 = {}, s1 = {};
    {
      const _Float16* k0p = Klds[cur] + ln * HEAD_DIM + g * 16;
      const _Float16* k1p = Klds[cur] + (16 + ln) * HEAD_DIM + g * 16;
      s0 = wmma_f16(aq0, load_b16(k0p), s0);
      s1 = wmma_f16(aq0, load_b16(k1p), s1);
      s0 = wmma_f16(aq1, load_b16(k0p + 32), s0);
      s1 = wmma_f16(aq1, load_b16(k1p + 32), s1);
    }

    // ---- online softmax update (row = vgpr + g*8, spread over 16 lanes) ---
#pragma unroll
    for (int v = 0; v < 8; ++v) {
      float x0 = s0[v] * ATT_SCALE;
      float x1 = s1[v] * ATT_SCALE;
      float mx = fmaxf(x0, x1);
#pragma unroll
      for (int msk = 1; msk < 16; msk <<= 1)
        mx = fmaxf(mx, __shfl_xor(mx, msk, 32));
      const float Mn = fmaxf(Mr[v], mx);
      const float al = __expf(Mr[v] - Mn);
      const float e0 = __expf(x0 - Mn);
      const float e1 = __expf(x1 - Mn);
      float rs = e0 + e1;
#pragma unroll
      for (int msk = 1; msk < 16; msk <<= 1)
        rs += __shfl_xor(rs, msk, 32);
      Lr[v] = Lr[v] * al + rs;
      Mr[v] = Mn;
      acc0[v] *= al; acc1[v] *= al; acc2[v] *= al; acc3[v] *= al;
      // stash P (f16) row-major 16x32 for the D->A re-layout
      Pw[(g * 8 + v) * 32 + ln]      = (_Float16)e0;
      Pw[(g * 8 + v) * 32 + 16 + ln] = (_Float16)e1;
    }
    __syncthreads();   // P visible; all waves done reading Klds[cur]

    // ---- P(16x32) @ V(32x64): 4 WMMAs, B tiles from LDS V ----------------
    const v16h ap = load_a16(Pw + ln * 32 + g * 16);
    {
      const _Float16* v0p = Vlds[cur] + ln * 32 + g * 16;
      acc0 = wmma_f16(ap, load_b16(v0p),            acc0);
      acc1 = wmma_f16(ap, load_b16(v0p + 16 * 32),  acc1);
      acc2 = wmma_f16(ap, load_b16(v0p + 32 * 32),  acc2);
      acc3 = wmma_f16(ap, load_b16(v0p + 48 * 32),  acc3);
    }
    __syncthreads();   // all waves done with this buffer before it is reused
  }

  // ---- normalize and scatter ctx back to [B,S,D_MODEL] layout -------------
#pragma unroll
  for (int v = 0; v < 8; ++v) {
    const float inv = 1.0f / Lr[v];
    const int srow = q0 + g * 8 + v;
    _Float16* crow = Ctx + ((size_t)bb * SEQ + srow) * D_MODEL + h * HEAD_DIM;
    crow[ln]      = (_Float16)(acc0[v] * inv);
    crow[16 + ln] = (_Float16)(acc1[v] * inv);
    crow[32 + ln] = (_Float16)(acc2[v] * inv);
    crow[48 + ln] = (_Float16)(acc3[v] * inv);
  }
}

// ---------------------------------------------------------------------------
// Kernel 4: output projection, fp32 result:  out = Ctx @ Wo^T + bo
// ---------------------------------------------------------------------------
__global__ void __launch_bounds__(256)
out_gemm(const _Float16* __restrict__ Ctx,
         const _Float16* __restrict__ Woh,
         const float* __restrict__ bo,
         float* __restrict__ out) {
  const int lane = threadIdx.x & 31;
  const int wid  = threadIdx.x >> 5;
  const int tile = blockIdx.x * 8 + wid;
  const int m0 = (tile >> 6) * 16;
  const int n0 = (tile & 63) * 16;
  const int g = lane >> 4, ln = lane & 15;

  const _Float16* Arow = Ctx + (size_t)(m0 + ln) * D_MODEL + g * 16;
  const _Float16* Brow = Woh + (size_t)(n0 + ln) * D_MODEL + g * 16;

  v8f c = {};
  for (int k0 = 0; k0 < D_MODEL; k0 += 32) {
    __builtin_prefetch(Arow + k0 + 128, 0, 3);
    c = wmma_f16(load_a16(Arow + k0), load_b16(Brow + k0), c);
  }

  const int e = n0 + ln;
  const float bias = bo[e];
#pragma unroll
  for (int v = 0; v < 8; ++v) {
    const int m = m0 + g * 8 + v;
    out[(size_t)m * D_MODEL + e] = c[v] + bias;
  }
}

// ---------------------------------------------------------------------------
// Launcher
// ---------------------------------------------------------------------------
extern "C" void kernel_launch(void* const* d_in, const int* in_sizes, int n_in,
                              void* d_out, int out_size, void* d_ws, size_t ws_size,
                              hipStream_t stream) {
  const float* x  = (const float*)d_in[0];
  const float* Wq = (const float*)d_in[1];
  const float* bq = (const float*)d_in[2];
  const float* Wk = (const float*)d_in[3];
  const float* bk = (const float*)d_in[4];
  const float* Wv = (const float*)d_in[5];
  const float* bv = (const float*)d_in[6];
  const float* Wo = (const float*)d_in[7];
  const float* bo = (const float*)d_in[8];

  char* ws = (char*)d_ws;
  const size_t MB = 1024 * 1024;
  _Float16* Xh  = (_Float16*)(ws);              // 16 MiB  (8192x1024 f16)
  _Float16* Wqh = (_Float16*)(ws + 16 * MB);    //  2 MiB
  _Float16* Wkh = (_Float16*)(ws + 18 * MB);
  _Float16* Wvh = (_Float16*)(ws + 20 * MB);
  _Float16* Woh = (_Float16*)(ws + 22 * MB);
  _Float16* Qh  = (_Float16*)(ws + 24 * MB);    // 16 MiB [B,H,S,64]
  _Float16* Kh  = (_Float16*)(ws + 40 * MB);    // 16 MiB [B,H,S,64]
  _Float16* Vt  = (_Float16*)(ws + 56 * MB);    // 16 MiB [B,H,64,S]
  _Float16* Ctx = (_Float16*)(ws + 72 * MB);    // 16 MiB [B,S,D]

  // 1) convert inputs to f16
  cvt_f32_to_f16<<<2048, 256, 0, stream>>>(x,  Xh,  NTOK * D_MODEL);
  cvt_f32_to_f16<<<1024, 256, 0, stream>>>(Wq, Wqh, D_MODEL * D_MODEL);
  cvt_f32_to_f16<<<1024, 256, 0, stream>>>(Wk, Wkh, D_MODEL * D_MODEL);
  cvt_f32_to_f16<<<1024, 256, 0, stream>>>(Wv, Wvh, D_MODEL * D_MODEL);
  cvt_f32_to_f16<<<1024, 256, 0, stream>>>(Wo, Woh, D_MODEL * D_MODEL);

  // 2) fused QKV projection: 512 x 64 tiles, 8 waves per block
  qkv_gemm<<<4096, 256, 0, stream>>>(Xh, Wqh, Wkh, Wvh, bq, bk, bv, Qh, Kh, Vt);

  // 3) flash attention: grid (B*H, S/64), 4 waves per block, double-buffered
  //    async K/V staging in LDS
  attn_kernel<<<dim3(BATCH * N_HEADS, SEQ / 64), 128, 0, stream>>>(Qh, Kh, Vt, Ctx);

  // 4) output projection (fp32 out + bias)
  out_gemm<<<4096, 256, 0, stream>>>(Ctx, Woh, bo, (float*)d_out);
}
